// ClusterMI_61168924230189
// MI455X (gfx1250) — matrix operand order repaired and checked by reading, not compiled, not run on hardware
//
#include <hip/hip_runtime.h>
#include <hip/hip_bf16.h>
#include <math.h>

// ---- CDNA5 WMMA vector types ----
typedef __attribute__((ext_vector_type(16))) _Float16     v16h;
typedef __attribute__((ext_vector_type(8)))  float        v8f;
typedef __attribute__((ext_vector_type(4)))  unsigned int v4u;

#define N_PTS      16384
#define N_DIM      128
#define NCLASS     10
#define KNN        3
#define BIGF       1.0e7f
#define EPSF       1e-8f
#define WAVES_PB   8                 // 256-thread blocks = 8 wave32s
#define ROW_TILES  (N_PTS / 16)
#define CHUNKS     4                 // column-space split for occupancy
#define CHUNK_COLS (N_PTS / CHUNKS)

// workspace layout (bytes); total ~5.2 MB
#define WS_XH      0                          // 16384*128 f16          = 4 MB
#define WS_ANCHOR  (N_PTS * N_DIM * 2)        // 16384 f32              = 64 KB
#define WS_DG      (WS_ANCHOR + N_PTS * 4)    // 16384 f32              = 64 KB
#define WS_COUNTS  (WS_DG + N_PTS * 4)        // 10 int (padded to 64B)
#define WS_TOP4    (WS_COUNTS + 64)           // CHUNKS*16384*4 f32     = 1 MB
#define WS_ROWCNT  (WS_TOP4 + CHUNKS * N_PTS * 4 * 4)  // 16384 int     = 64 KB

// ---------------- digamma (psi), f32 accuracy ------------------------------
__device__ __forceinline__ float dev_digamma(float x) {
    float r = 0.0f;
    while (x < 6.0f) { r -= 1.0f / x; x += 1.0f; }
    float inv = 1.0f / x, inv2 = inv * inv;
    return r + logf(x) - 0.5f * inv
           - inv2 * (1.0f / 12.0f - inv2 * (1.0f / 120.0f - inv2 * (1.0f / 252.0f)));
}

// ---- top-4 insertion, compare-select (no NaN canonicalize ops) ------------
__device__ __forceinline__ void ins4(float v, float& t0, float& t1,
                                     float& t2, float& t3) {
    float a;
    a = (t0 >= v) ? t0 : v; v = (t0 >= v) ? v : t0; t0 = a;
    a = (t1 >= v) ? t1 : v; v = (t1 >= v) ? v : t1; t1 = a;
    a = (t2 >= v) ? t2 : v; v = (t2 >= v) ? v : t2; t2 = a;
    t3 = (t3 >= v) ? t3 : v;
}

// ---------------- row-normalize X -> f16, one wave per row -----------------
__global__ __launch_bounds__(256) void k_normalize(const float* __restrict__ X,
                                                   _Float16* __restrict__ Xh) {
    const int wv   = threadIdx.x >> 5;
    const int lane = threadIdx.x & 31;
    const int row  = blockIdx.x * WAVES_PB + wv;
    const float4 v = *(const float4*)(X + (size_t)row * N_DIM + lane * 4);
    float ss = v.x * v.x + v.y * v.y + v.z * v.z + v.w * v.w;
#pragma unroll
    for (int off = 16; off > 0; off >>= 1) ss += __shfl_xor(ss, off, 32);
    const float inv = 1.0f / fmaxf(sqrtf(ss), EPSF);
    union { _Float16 h[4]; uint2 u; } o;
    o.h[0] = (_Float16)(v.x * inv); o.h[1] = (_Float16)(v.y * inv);
    o.h[2] = (_Float16)(v.z * inv); o.h[3] = (_Float16)(v.w * inv);
    *(uint2*)(Xh + (size_t)row * N_DIM + lane * 4) = o.u;
}

// ---------------- fragment loaders (contiguous k = 16h + s, same for A and B) -----
// A 16x32 f16: lanes 0-15 M=lane, lanes 16-31 M=lane-16; half=lane/16 picks
// K = k0+16h .. +15 contiguous (matches B): two b128 loads.
__device__ __forceinline__ v16h load_a_frag(const _Float16* __restrict__ rowp,
                                            int k0, int half) {
    union { v16h v; v4u q[2]; } u;
    u.q[0] = *(const v4u*)(rowp + k0 + 16 * half);
    u.q[1] = *(const v4u*)(rowp + k0 + 16 * half + 8);
    return u.v;
}
// B 32x16 f16 (B = Xh^T): N=lane%16, K = k0+16*half .. +15 contiguous.
__device__ __forceinline__ v16h load_b_frag(const _Float16* __restrict__ colp,
                                            int k0) {
    union { v16h v; v4u q[2]; } u;
    u.q[0] = *(const v4u*)(colp + k0);
    u.q[1] = *(const v4u*)(colp + k0 + 8);
    return u.v;
}

// two 16x16 tiles, interleaved accumulator chains (hides WMMA hazards)
__device__ __forceinline__ void tile_mm2(const v16h a[4],
                                         const _Float16* __restrict__ Bc0,
                                         const _Float16* __restrict__ Bc1,
                                         v8f& acc0, v8f& acc1) {
    v16h b0[4], b1[4];
#pragma unroll
    for (int k = 0; k < 4; ++k) b0[k] = load_b_frag(Bc0, k * 32);
#pragma unroll
    for (int k = 0; k < 4; ++k) b1[k] = load_b_frag(Bc1, k * 32);
    v8f c0 = {}, c1 = {};
#pragma unroll
    for (int k = 0; k < 4; ++k) {
        c0 = __builtin_amdgcn_wmma_f32_16x16x32_f16(false, a[k], false, b0[k],
                                                    (short)0, c0, false, false);
        c1 = __builtin_amdgcn_wmma_f32_16x16x32_f16(false, a[k], false, b1[k],
                                                    (short)0, c1, false, false);
    }
    acc0 = c0; acc1 = c1;
}

// ---------------- pass A: per-chunk top-4 same-class similarities ----------
__global__ __launch_bounds__(256) void k_anchor_partial(
        const _Float16* __restrict__ Xh, const int* __restrict__ y,
        float* __restrict__ top4c) {
    __shared__ float lred[WAVES_PB][16][16][4];   // 32 KB
    const int wv   = threadIdx.x >> 5;
    const int lane = threadIdx.x & 31;
    const int n    = lane & 15;
    const int half = lane >> 4;
    // wave-uniform -> tell the compiler (scalar branches / scalar addressing)
    const int row0 = __builtin_amdgcn_readfirstlane(
                        (blockIdx.x * WAVES_PB + wv) * 16);
    const int cbeg = blockIdx.y * CHUNK_COLS;

    // whole A row-panel (16 rows x 128 K) resident in 32 VGPRs
    const _Float16* Ar = Xh + (size_t)(row0 + n) * N_DIM;
    v16h a[4];
#pragma unroll
    for (int k = 0; k < 4; ++k) a[k] = load_a_frag(Ar, k * 32, half);

    int yrow[8];
#pragma unroll
    for (int r = 0; r < 8; ++r) yrow[r] = y[row0 + 8 * half + r];

    float t0[8], t1[8], t2[8], t3[8];
#pragma unroll
    for (int r = 0; r < 8; ++r) { t0[r] = t1[r] = t2[r] = t3[r] = -3.0e38f; }

    for (int c0 = 0; c0 < CHUNK_COLS; c0 += 32) {
        const int ca = cbeg + c0, cb = ca + 16;
        const _Float16* Bc0 = Xh + (size_t)(ca + n) * N_DIM + 16 * half;
        const _Float16* Bc1 = Bc0 + (size_t)16 * N_DIM;
        __builtin_prefetch(Bc0 + N_DIM * 32, 0, 1);   // next iter, tile 0
        __builtin_prefetch(Bc1 + N_DIM * 32, 0, 1);   // next iter, tile 1
        v8f acc0, acc1;
        tile_mm2(a, Bc0, Bc1, acc0, acc1);

        const int yc0 = y[ca + n], yc1 = y[cb + n];
        float va[8], vb[8];
#pragma unroll
        for (int r = 0; r < 8; ++r) {
            va[r] = (yc0 == yrow[r]) ? acc0[r] : -BIGF;
            vb[r] = (yc1 == yrow[r]) ? acc1[r] : -BIGF;
        }
        if (ca == row0) {                        // scalar branch: diag in tile0
            const int col = ca + n;
#pragma unroll
            for (int r = 0; r < 8; ++r)
                if (col == row0 + 8 * half + r) va[r] = 1.0f;   // exact self
        }
        if (cb == row0) {                        // scalar branch: diag in tile1
            const int col = cb + n;
#pragma unroll
            for (int r = 0; r < 8; ++r)
                if (col == row0 + 8 * half + r) vb[r] = 1.0f;
        }
#pragma unroll
        for (int r = 0; r < 8; ++r) {
            ins4(va[r], t0[r], t1[r], t2[r], t3[r]);
            ins4(vb[r], t0[r], t1[r], t2[r], t3[r]);
        }
    }

#pragma unroll
    for (int r = 0; r < 8; ++r) {
        const int rl = 8 * half + r;
        lred[wv][rl][n][0] = t0[r]; lred[wv][rl][n][1] = t1[r];
        lred[wv][rl][n][2] = t2[r]; lred[wv][rl][n][3] = t3[r];
    }
    __syncthreads();
    if (lane < 16) {
        float s0 = -3.0e38f, s1 = s0, s2 = s0, s3 = s0;
        for (int j = 0; j < 16; ++j)
#pragma unroll
            for (int k = 0; k < 4; ++k)
                ins4(lred[wv][lane][j][k], s0, s1, s2, s3);
        float4* dst = (float4*)(top4c +
                      ((size_t)blockIdx.y * N_PTS + row0 + lane) * 4);
        *dst = make_float4(s0, s1, s2, s3);
    }
}

// merge per-chunk top-4 -> global anchor (4th largest incl. self)
__global__ __launch_bounds__(256) void k_anchor_merge(
        const float* __restrict__ top4c, float* __restrict__ anchor) {
    const int row = blockIdx.x * blockDim.x + threadIdx.x;
    float t0 = -3.0e38f, t1 = t0, t2 = t0, t3 = t0;
#pragma unroll
    for (int c = 0; c < CHUNKS; ++c) {
        const float4 q = *(const float4*)(top4c + ((size_t)c * N_PTS + row) * 4);
        ins4(q.x, t0, t1, t2, t3); ins4(q.y, t0, t1, t2, t3);
        ins4(q.z, t0, t1, t2, t3); ins4(q.w, t0, t1, t2, t3);
    }
    anchor[row] = t3;
}

// ---------------- pass B: partial counts of { sim >= anchor } --------------
__global__ __launch_bounds__(256) void k_count_partial(
        const _Float16* __restrict__ Xh, const float* __restrict__ anchor,
        int* __restrict__ rowcnt) {
    __shared__ int lcnt[WAVES_PB][16][16];   // 8 KB
    const int wv   = threadIdx.x >> 5;
    const int lane = threadIdx.x & 31;
    const int n    = lane & 15;
    const int half = lane >> 4;
    const int row0 = __builtin_amdgcn_readfirstlane(
                        (blockIdx.x * WAVES_PB + wv) * 16);
    const int cbeg = blockIdx.y * CHUNK_COLS;

    const _Float16* Ar = Xh + (size_t)(row0 + n) * N_DIM;
    v16h a[4];
#pragma unroll
    for (int k = 0; k < 4; ++k) a[k] = load_a_frag(Ar, k * 32, half);

    float arow[8];
#pragma unroll
    for (int r = 0; r < 8; ++r) arow[r] = anchor[row0 + 8 * half + r];

    int cnt[8] = {0, 0, 0, 0, 0, 0, 0, 0};

    for (int c0 = 0; c0 < CHUNK_COLS; c0 += 32) {
        const int ca = cbeg + c0, cb = ca + 16;
        const _Float16* Bc0 = Xh + (size_t)(ca + n) * N_DIM + 16 * half;
        const _Float16* Bc1 = Bc0 + (size_t)16 * N_DIM;
        __builtin_prefetch(Bc0 + N_DIM * 32, 0, 1);
        __builtin_prefetch(Bc1 + N_DIM * 32, 0, 1);
        v8f acc0, acc1;
        tile_mm2(a, Bc0, Bc1, acc0, acc1);

        float va[8], vb[8];
#pragma unroll
        for (int r = 0; r < 8; ++r) { va[r] = acc0[r]; vb[r] = acc1[r]; }
        if (ca == row0) {                        // scalar branch: diag fix-up
            const int col = ca + n;
#pragma unroll
            for (int r = 0; r < 8; ++r)
                if (col == row0 + 8 * half + r) va[r] = 1.0f;
        }
        if (cb == row0) {
            const int col = cb + n;
#pragma unroll
            for (int r = 0; r < 8; ++r)
                if (col == row0 + 8 * half + r) vb[r] = 1.0f;
        }
#pragma unroll
        for (int r = 0; r < 8; ++r) {
            cnt[r] += (va[r] >= arow[r]) ? 1 : 0;   // dist<=anchor <=> sim>=anchor
            cnt[r] += (vb[r] >= arow[r]) ? 1 : 0;
        }
    }

#pragma unroll
    for (int r = 0; r < 8; ++r) lcnt[wv][8 * half + r][n] = cnt[r];
    __syncthreads();
    if (lane < 16) {
        int s = 0;
        for (int j = 0; j < 16; ++j) s += lcnt[wv][lane][j];
        atomicAdd(&rowcnt[row0 + lane], s);   // int atomics: deterministic
    }
}

// rowcnt -> digamma(m_i), m_i = count - 1 (remove self)
__global__ __launch_bounds__(256) void k_digamma(const int* __restrict__ rowcnt,
                                                 float* __restrict__ dg) {
    const int i = blockIdx.x * blockDim.x + threadIdx.x;
    dg[i] = dev_digamma((float)(rowcnt[i] - 1));
}

// ---------------- tiny helpers --------------------------------------------
__global__ __launch_bounds__(256) void k_init(int* counts, int* rowcnt) {
    const int i = blockIdx.x * blockDim.x + threadIdx.x;
    rowcnt[i] = 0;
    if (i < NCLASS) counts[i] = 0;
}
__global__ void k_class_counts(const int* __restrict__ y, int* counts) {
    const int i = blockIdx.x * blockDim.x + threadIdx.x;
    atomicAdd(&counts[y[i]], 1);
}
__global__ __launch_bounds__(256) void k_final(const float* __restrict__ dg,
                                               const int* __restrict__ counts,
                                               float* __restrict__ out) {
    __shared__ float red[256];
    float s = 0.0f;
    for (int i = threadIdx.x; i < N_PTS; i += 256) s += dg[i];  // fixed order
    red[threadIdx.x] = s;
    __syncthreads();
    for (int off = 128; off > 0; off >>= 1) {
        if (threadIdx.x < off) red[threadIdx.x] += red[threadIdx.x + off];
        __syncthreads();
    }
    if (threadIdx.x == 0) {
        const float avg_m = red[0] / (float)N_PTS;
        float avgNx = 0.0f;
        for (int c = 0; c < NCLASS; ++c) {
            const float cf = (float)counts[c];
            if (cf > 0.0f) avgNx += (cf / (float)N_PTS) * dev_digamma(cf);
        }
        const float mi = dev_digamma((float)N_PTS) - avgNx
                       + dev_digamma((float)KNN) - avg_m;
        out[0] = mi / logf(2.0f);
    }
}

// ---------------- launcher --------------------------------------------------
extern "C" void kernel_launch(void* const* d_in, const int* in_sizes, int n_in,
                              void* d_out, int out_size, void* d_ws, size_t ws_size,
                              hipStream_t stream) {
    const float* X = (const float*)d_in[0];
    const int*   y = (const int*)d_in[1];
    float*     out = (float*)d_out;
    char*       ws = (char*)d_ws;

    _Float16* Xh     = (_Float16*)(ws + WS_XH);
    float*    anchor = (float*)(ws + WS_ANCHOR);
    float*    dg     = (float*)(ws + WS_DG);
    int*      counts = (int*)(ws + WS_COUNTS);
    float*    top4c  = (float*)(ws + WS_TOP4);
    int*      rowcnt = (int*)(ws + WS_ROWCNT);

    const dim3 gemmGrid(ROW_TILES / WAVES_PB, CHUNKS);

    k_init          <<<N_PTS / 256, 256, 0, stream>>>(counts, rowcnt);
    k_normalize     <<<N_PTS / WAVES_PB, 256, 0, stream>>>(X, Xh);
    k_class_counts  <<<N_PTS / 256, 256, 0, stream>>>(y, counts);
    k_anchor_partial<<<gemmGrid, 256, 0, stream>>>(Xh, y, top4c);
    k_anchor_merge  <<<N_PTS / 256, 256, 0, stream>>>(top4c, anchor);
    k_count_partial <<<gemmGrid, 256, 0, stream>>>(Xh, anchor, rowcnt);
    k_digamma       <<<N_PTS / 256, 256, 0, stream>>>(rowcnt, dg);
    k_final         <<<1, 256, 0, stream>>>(dg, counts, out);
}
